// EGNNQM9PerturbRegressionTarget_30983894073592
// MI455X (gfx1250) — matrix operand order repaired and checked by reading, not compiled, not run on hardware
//
#include <hip/hip_runtime.h>

typedef __attribute__((ext_vector_type(16))) _Float16 v16h;
typedef __attribute__((ext_vector_type(8)))  _Float16 v8h;
typedef __attribute__((ext_vector_type(8)))  float    v8f;

#define HID 128

__device__ __forceinline__ v8f wmma16(v16h a, v16h b, v8f c) {
  // D = A(16x32 f16) * B(32x16 f16) + C(16x16 f32)
  return __builtin_amdgcn_wmma_f32_16x16x32_f16(false, a, false, b, (short)0, c, false, false);
}
__device__ __forceinline__ float siluf(float v) { return v / (1.0f + __expf(-v)); }
__device__ __forceinline__ float sigmf(float v) { return 1.0f / (1.0f + __expf(-v)); }

__device__ __forceinline__ v8h cvt8(const float* __restrict__ p) {
  float4 a = ((const float4*)p)[0];
  float4 b = ((const float4*)p)[1];
  v8h d;
  d[0]=(_Float16)a.x; d[1]=(_Float16)a.y; d[2]=(_Float16)a.z; d[3]=(_Float16)a.w;
  d[4]=(_Float16)b.x; d[5]=(_Float16)b.y; d[6]=(_Float16)b.z; d[7]=(_Float16)b.w;
  return d;
}
__device__ __forceinline__ v16h cat8(v8h lo, v8h hh) {
  v16h a;
#pragma unroll
  for (int i = 0; i < 8; ++i) { a[i] = lo[i]; a[i+8] = hh[i]; }
  return a;
}
__device__ __forceinline__ void atomAddF(float* p, float v) {
  __hip_atomic_fetch_add(p, v, __ATOMIC_RELAXED, __HIP_MEMORY_SCOPE_AGENT);
}

// ---------------------------------------------------------------------------
// Pack f32 row-major weights [Lc][rows_src][128] into WMMA B-fragment order:
// dst[((l*nK + ks)*8 + nt)*32 + lane][16 halfs], lane: n = nt*16 + (lane&15),
// k = ks*32 + (i>>3)*16 + (lane>>4)*8 + (i&7). Rows >= rows_src are zero (K padding).
// ---------------------------------------------------------------------------
__global__ void egnn_pack_b(const float* __restrict__ src, _Float16* __restrict__ dst,
                            int Lc, int rows_src, int nK) {
  long idx = (long)blockIdx.x * blockDim.x + threadIdx.x;
  long total = (long)Lc * nK * 8 * 32 * 16;
  if (idx >= total) return;
  int i    = (int)(idx & 15);
  long t   = idx >> 4;
  int lane = (int)(t & 31); t >>= 5;
  int nt   = (int)(t & 7);  t >>= 3;
  int ks   = (int)(t % nK);
  int l    = (int)(t / nK);
  int n = nt * 16 + (lane & 15);
  int k = ks * 32 + (i >> 3) * 16 + (lane >> 4) * 8 + (i & 7);
  float v = (k < rows_src) ? src[((long)l * rows_src + k) * HID + n] : 0.0f;
  dst[idx] = (_Float16)v;
}

__global__ void egnn_zero(float* __restrict__ p, int n) {
  int i = blockIdx.x * blockDim.x + threadIdx.x;
  if (i < n) p[i] = 0.0f;
}

__global__ void egnn_axpy(float* __restrict__ x, const float* __restrict__ a, int n) {
  int i = blockIdx.x * blockDim.x + threadIdx.x;
  if (i < n) x[i] += a[i];
}

// h = x_feat @ emb_W + emb_b   [N,11] @ [11,128]
__global__ void egnn_emb(const float* __restrict__ xf, const float* __restrict__ W,
                         const float* __restrict__ b, float* __restrict__ h, int N) {
  int idx = blockIdx.x * blockDim.x + threadIdx.x;
  if (idx >= N * HID) return;
  int n = idx >> 7, c = idx & 127;
  float s = b[c];
#pragma unroll
  for (int k = 0; k < 11; ++k) s += xf[n * 11 + k] * W[k * HID + c];
  h[idx] = s;
}

// ---------------------------------------------------------------------------
// Edge kernel: one wave handles 32 edges (two 16-row M-tiles) so every B
// fragment load feeds two WMMAs.
//   m1 = silu(msg_in @ W1 + b1)            K = 288 (261 padded)
//   m  = silu(m1 @ W2 + b2)
//   m *= sigmoid(m @ gW + gb) * mask
//   m_aggr[col] += m   (atomics)
//   cw = silu(m @ Wc1 + bc1) @ Wc2 + cb2
//   x_acc[col] += coord_diff * cw          (atomics)
// ---------------------------------------------------------------------------
__global__ __launch_bounds__(128) void egnn_edge(
    const float* __restrict__ h, const float* __restrict__ x,
    const float* __restrict__ eattr, const float* __restrict__ p_in,
    const int* __restrict__ erow, const int* __restrict__ ecol,
    const int* __restrict__ einv,
    const v16h* __restrict__ W1pk, const float* __restrict__ b1,
    const v16h* __restrict__ W2pk, const float* __restrict__ b2,
    const float* __restrict__ gW, const float* __restrict__ gb,
    const v16h* __restrict__ Wc1pk, const float* __restrict__ bc1,
    const float* __restrict__ Wc2, const float* __restrict__ cb2,
    float* __restrict__ m_aggr, float* __restrict__ x_acc,
    int nTiles, int E)
{
  __shared__ _Float16 lds_all[4 * 32 * HID];          // 4 waves x 8KB
  const int lane = threadIdx.x & 31;
  const int wib  = threadIdx.x >> 5;
  const int tile = blockIdx.x * 4 + wib;              // 32-edge tile
  if (tile >= nTiles) return;
  _Float16* Lb = lds_all + wib * (32 * HID);
  const int mrow = lane & 15;
  const int hi   = lane >> 4;
  const int nl   = mrow;                              // column index in C layout
  const int e0   = tile * 32;

  // per-M-tile edge scalars
  int ee[2];
  const float* hc[2];
  const float* hr[2];
  float dist[2];
#pragma unroll
  for (int t = 0; t < 2; ++t) {
    int e = e0 + t * 16 + mrow; if (e >= E) e = E - 1;
    ee[t] = e;
    const int ri = erow[e], ci = ecol[e];
    hc[t] = h + (size_t)ci * HID;
    hr[t] = h + (size_t)ri * HID;
    const float dx = x[ci*3+0] - x[ri*3+0];
    const float dy = x[ci*3+1] - x[ri*3+1];
    const float dz = x[ci*3+2] - x[ri*3+2];
    dist[t] = sqrtf(dx*dx + dy*dy + dz*dz);
  }

  const v8f z8 = {0,0,0,0,0,0,0,0};

  // ---- matmul 1: msg_in[32,288] @ W1, B fragment shared by both M-tiles ----
  v8f acc[2][8];
#pragma unroll
  for (int t = 0; t < 2; ++t)
#pragma unroll
    for (int nt = 0; nt < 8; ++nt) acc[t][nt] = z8;
#pragma unroll
  for (int ks = 0; ks < 9; ++ks) {
    v16h a[2];
#pragma unroll
    for (int t = 0; t < 2; ++t) {
      if (ks < 8) {
        const float* src = (ks < 4) ? hc[t] : hr[t];   // [h_col | h_row]
        const int o = (ks & 3) * 32 + hi * 8;
        a[t] = cat8(cvt8(src + o), cvt8(src + o + 16));
      } else {                                         // K=256..287: dist, edge_attr, pad
#pragma unroll
        for (int i = 0; i < 16; ++i) a[t][i] = (_Float16)0.0f;
        if (hi == 0) {
          a[t][0] = (_Float16)dist[t];
          const float* ea = eattr + (size_t)ee[t] * 4;
          a[t][1] = (_Float16)ea[0]; a[t][2] = (_Float16)ea[1];
          a[t][3] = (_Float16)ea[2]; a[t][4] = (_Float16)ea[3];
        }
      }
    }
#pragma unroll
    for (int nt = 0; nt < 8; ++nt) {
      const v16h b = W1pk[(ks * 8 + nt) * 32 + lane];
      acc[0][nt] = wmma16(a[0], b, acc[0][nt]);
      acc[1][nt] = wmma16(a[1], b, acc[1][nt]);
    }
  }
  // bias + silu -> LDS (C layout -> row-major f16 tiles for next A fragments)
#pragma unroll
  for (int t = 0; t < 2; ++t)
#pragma unroll
    for (int nt = 0; nt < 8; ++nt) {
      const float bias = b1[nt * 16 + nl];
#pragma unroll
      for (int r = 0; r < 8; ++r)
        Lb[(t*16 + r + hi*8) * HID + nt*16 + nl] = (_Float16)siluf(acc[t][nt][r] + bias);
    }

  // ---- matmul 2: m = silu(m1 @ W2 + b2), ks-outer (A loaded once per ks) ----
  v8f m[2][8];
#pragma unroll
  for (int t = 0; t < 2; ++t)
#pragma unroll
    for (int nt = 0; nt < 8; ++nt) m[t][nt] = z8;
#pragma unroll
  for (int ks = 0; ks < 4; ++ks) {
    const int o = ks * 32 + hi * 8;
    v16h a[2];
#pragma unroll
    for (int t = 0; t < 2; ++t)
      a[t] = cat8(*(const v8h*)(Lb + (t*16 + mrow) * HID + o),
                  *(const v8h*)(Lb + (t*16 + mrow) * HID + o + 16));
#pragma unroll
    for (int nt = 0; nt < 8; ++nt) {
      const v16h b = W2pk[(ks * 8 + nt) * 32 + lane];
      m[0][nt] = wmma16(a[0], b, m[0][nt]);
      m[1][nt] = wmma16(a[1], b, m[1][nt]);
    }
  }
#pragma unroll
  for (int t = 0; t < 2; ++t)
#pragma unroll
    for (int nt = 0; nt < 8; ++nt) {
      const float bias = b2[nt * 16 + nl];
#pragma unroll
      for (int r = 0; r < 8; ++r) m[t][nt][r] = siluf(m[t][nt][r] + bias);
    }

  // ---- gate: per-edge dot over channels, 16-lane butterfly ----
  float part[2][8];
#pragma unroll
  for (int t = 0; t < 2; ++t)
#pragma unroll
    for (int r = 0; r < 8; ++r) part[t][r] = 0.0f;
#pragma unroll
  for (int nt = 0; nt < 8; ++nt) {
    const float w = gW[nt * 16 + nl];
#pragma unroll
    for (int t = 0; t < 2; ++t)
#pragma unroll
      for (int r = 0; r < 8; ++r) part[t][r] += m[t][nt][r] * w;
  }
#pragma unroll
  for (int o = 1; o < 16; o <<= 1)
#pragma unroll
    for (int t = 0; t < 2; ++t)
#pragma unroll
      for (int r = 0; r < 8; ++r) part[t][r] += __shfl_xor(part[t][r], o, 32);

  const float gbv = gb[0];
  int ecols[2][8];
#pragma unroll
  for (int t = 0; t < 2; ++t)
#pragma unroll
    for (int r = 0; r < 8; ++r) {
      int em = e0 + t*16 + r + hi*8;
      const bool ok = em < E;
      if (em >= E) em = E - 1;
      ecols[t][r] = ecol[em];
      const float gate = sigmf(part[t][r] + gbv);
      const float mask = (p_in[einv[em]] >= 0.0f) ? 1.0f : 0.0f;  // straight-through fwd
      const float s = ok ? gate * mask : 0.0f;
#pragma unroll
      for (int nt = 0; nt < 8; ++nt) m[t][nt][r] *= s;
    }

  // ---- masked m -> LDS for coord MLP ----
#pragma unroll
  for (int t = 0; t < 2; ++t)
#pragma unroll
    for (int nt = 0; nt < 8; ++nt)
#pragma unroll
      for (int r = 0; r < 8; ++r)
        Lb[(t*16 + r + hi*8) * HID + nt*16 + nl] = (_Float16)m[t][nt][r];

  // ---- cw = silu(m @ Wc1 + bc1) @ Wc2 + cb2, ks-outer ----
  v8f cacc[2][8];
#pragma unroll
  for (int t = 0; t < 2; ++t)
#pragma unroll
    for (int nt = 0; nt < 8; ++nt) cacc[t][nt] = z8;
#pragma unroll
  for (int ks = 0; ks < 4; ++ks) {
    const int o = ks * 32 + hi * 8;
    v16h a[2];
#pragma unroll
    for (int t = 0; t < 2; ++t)
      a[t] = cat8(*(const v8h*)(Lb + (t*16 + mrow) * HID + o),
                  *(const v8h*)(Lb + (t*16 + mrow) * HID + o + 16));
#pragma unroll
    for (int nt = 0; nt < 8; ++nt) {
      const v16h b = Wc1pk[(ks * 8 + nt) * 32 + lane];
      cacc[0][nt] = wmma16(a[0], b, cacc[0][nt]);
      cacc[1][nt] = wmma16(a[1], b, cacc[1][nt]);
    }
  }
  float pc[2][8];
#pragma unroll
  for (int t = 0; t < 2; ++t)
#pragma unroll
    for (int r = 0; r < 8; ++r) pc[t][r] = 0.0f;
#pragma unroll
  for (int nt = 0; nt < 8; ++nt) {
    const float bias = bc1[nt * 16 + nl];
    const float w = Wc2[nt * 16 + nl];
#pragma unroll
    for (int t = 0; t < 2; ++t)
#pragma unroll
      for (int r = 0; r < 8; ++r) pc[t][r] += siluf(cacc[t][nt][r] + bias) * w;
  }
#pragma unroll
  for (int o = 1; o < 16; o <<= 1)
#pragma unroll
    for (int t = 0; t < 2; ++t)
#pragma unroll
      for (int r = 0; r < 8; ++r) pc[t][r] += __shfl_xor(pc[t][r], o, 32);

  // ---- scatter m into m_aggr[col] (lane-contiguous channels -> coalesced atomics) ----
#pragma unroll
  for (int t = 0; t < 2; ++t)
#pragma unroll
    for (int r = 0; r < 8; ++r) {
      const int em = e0 + t*16 + r + hi*8;
      if (em < E) {
        float* dst = m_aggr + (size_t)ecols[t][r] * HID + nl;
#pragma unroll
        for (int nt = 0; nt < 8; ++nt) atomAddF(dst + nt * 16, m[t][nt][r]);
      }
    }

  // ---- x_acc[col] += coord_diff * cw : lanes nl<8 handle edge e0+t*16+nl+hi*8 ----
  if (nl < 8) {
#pragma unroll
    for (int t = 0; t < 2; ++t) {
      const int me = e0 + t*16 + nl + hi*8;
      if (me < E) {
        float cw = pc[t][0];
#pragma unroll
        for (int r = 1; r < 8; ++r) cw = (nl == r) ? pc[t][r] : cw;
        cw += cb2[0];
        const int r2 = erow[me], c2 = ecol[me];
        const float ddx = x[c2*3+0] - x[r2*3+0];
        const float ddy = x[c2*3+1] - x[r2*3+1];
        const float ddz = x[c2*3+2] - x[r2*3+2];
        atomAddF(&x_acc[c2*3+0], ddx * cw);
        atomAddF(&x_acc[c2*3+1], ddy * cw);
        atomAddF(&x_acc[c2*3+2], ddz * cw);
      }
    }
  }
}

// ---------------------------------------------------------------------------
// Node kernel: one wave = 16 nodes. h += silu([h|m_aggr] @ Wn1 + bn1) @ Wn2 + bn2
// ---------------------------------------------------------------------------
__global__ __launch_bounds__(256) void egnn_node(
    float* __restrict__ h, const float* __restrict__ maggr,
    const v16h* __restrict__ W1pk, const float* __restrict__ b1,
    const v16h* __restrict__ W2pk, const float* __restrict__ b2,
    int N, int nTiles)
{
  __shared__ _Float16 lds_all[8 * 16 * HID];
  const int lane = threadIdx.x & 31;
  const int wib  = threadIdx.x >> 5;
  const int tile = blockIdx.x * 8 + wib;
  if (tile >= nTiles) return;
  _Float16* Lb = lds_all + wib * (16 * HID);
  const int mrow = lane & 15;
  const int hi = lane >> 4;
  const int nl = mrow;
  const int n0 = tile * 16;
  int node = n0 + mrow; if (node >= N) node = N - 1;
  const float* hn = h + (size_t)node * HID;
  const float* mn = maggr + (size_t)node * HID;

  const v8f z8 = {0,0,0,0,0,0,0,0};
  v8f acc[8];
#pragma unroll
  for (int nt = 0; nt < 8; ++nt) acc[nt] = z8;
#pragma unroll
  for (int ks = 0; ks < 8; ++ks) {                    // K = 256: [h | m_aggr]
    const float* src = (ks < 4) ? hn : mn;
    const int o = (ks & 3) * 32 + hi * 8;
    v16h a = cat8(cvt8(src + o), cvt8(src + o + 16));
#pragma unroll
    for (int nt = 0; nt < 8; ++nt)
      acc[nt] = wmma16(a, W1pk[(ks * 8 + nt) * 32 + lane], acc[nt]);
  }
#pragma unroll
  for (int nt = 0; nt < 8; ++nt) {
    const float bias = b1[nt * 16 + nl];
#pragma unroll
    for (int r = 0; r < 8; ++r)
      Lb[(r + hi * 8) * HID + nt * 16 + nl] = (_Float16)siluf(acc[nt][r] + bias);
  }
  // matmul 2, ks-outer
  v8f c2[8];
#pragma unroll
  for (int nt = 0; nt < 8; ++nt) c2[nt] = z8;
#pragma unroll
  for (int ks = 0; ks < 4; ++ks) {
    const int o = ks * 32 + hi * 8;
    v16h a = cat8(*(const v8h*)(Lb + mrow * HID + o),
                  *(const v8h*)(Lb + mrow * HID + o + 16));
#pragma unroll
    for (int nt = 0; nt < 8; ++nt)
      c2[nt] = wmma16(a, W2pk[(ks * 8 + nt) * 32 + lane], c2[nt]);
  }
#pragma unroll
  for (int nt = 0; nt < 8; ++nt) {
    const float bias = b2[nt * 16 + nl];
#pragma unroll
    for (int r = 0; r < 8; ++r) {
      const int nd = n0 + r + hi * 8;
      if (nd < N) h[(size_t)nd * HID + nt * 16 + nl] += c2[nt][r] + bias;  // unique owner
    }
  }
}

__global__ void egnn_pool(const float* __restrict__ h, const int* __restrict__ batch,
                          float* __restrict__ sums, float* __restrict__ counts, int N) {
  int idx = blockIdx.x * blockDim.x + threadIdx.x;
  if (idx >= N * HID) return;
  int n = idx >> 7, c = idx & 127;
  int g = batch[n];
  atomAddF(&sums[(size_t)g * HID + c], h[idx]);
  if (c == 0) atomAddF(&counts[g], 1.0f);
}

__global__ void egnn_readout(const float* __restrict__ sums, const float* __restrict__ counts,
                             const float* __restrict__ rW, const float* __restrict__ rb,
                             float* __restrict__ out, int G) {
  int g = blockIdx.x * blockDim.x + threadIdx.x;
  if (g >= G) return;
  float inv = 1.0f / fmaxf(counts[g], 1.0f);
  float s = 0.0f;
#pragma unroll 8
  for (int c = 0; c < HID; ++c) s += sums[(size_t)g * HID + c] * inv * rW[c];
  out[g] = s + rb[0];
}

// ---------------------------------------------------------------------------
extern "C" void kernel_launch(void* const* d_in, const int* in_sizes, int n_in,
                              void* d_out, int out_size, void* d_ws, size_t ws_size,
                              hipStream_t stream) {
  const float* x_feat = (const float*)d_in[0];
  const float* pos    = (const float*)d_in[1];
  const float* eattr  = (const float*)d_in[2];
  const float* p_in   = (const float*)d_in[3];
  const float* emb_W  = (const float*)d_in[4];
  const float* emb_b  = (const float*)d_in[5];
  const float* eW1 = (const float*)d_in[6];
  const float* eb1 = (const float*)d_in[7];
  const float* eW2 = (const float*)d_in[8];
  const float* eb2 = (const float*)d_in[9];
  const float* gW  = (const float*)d_in[10];
  const float* gb  = (const float*)d_in[11];
  const float* nW1 = (const float*)d_in[12];
  const float* nb1 = (const float*)d_in[13];
  const float* nW2 = (const float*)d_in[14];
  const float* nb2 = (const float*)d_in[15];
  const float* cW1 = (const float*)d_in[16];
  const float* cb1 = (const float*)d_in[17];
  const float* cW2 = (const float*)d_in[18];
  const float* cb2 = (const float*)d_in[19];
  const float* rW  = (const float*)d_in[20];
  const float* rb  = (const float*)d_in[21];
  const int* eidx  = (const int*)d_in[22];
  const int* einv  = (const int*)d_in[23];
  const int* batch = (const int*)d_in[24];

  const int N = in_sizes[1] / 3;
  const int E = in_sizes[22] / 2;
  const int G = out_size;
  const int Lnum = in_sizes[11];          // gb is [L,1]

  char* ws = (char*)d_ws;
  size_t off = 0;
  auto alloc = [&](size_t bytes) -> void* {
    void* r = ws + off;
    off += (bytes + 255) & ~(size_t)255;
    return r;
  };
  float* h      = (float*)alloc((size_t)N * HID * 4);
  float* maggr  = (float*)alloc((size_t)N * HID * 4);
  float* xb     = (float*)alloc((size_t)N * 3 * 4);
  float* xacc   = (float*)alloc((size_t)N * 3 * 4);
  float* sums   = (float*)alloc((size_t)G * HID * 4);
  float* counts = (float*)alloc((size_t)G * 4);
  _Float16* W1pk  = (_Float16*)alloc((size_t)Lnum * 9 * 4096 * 2);
  _Float16* W2pk  = (_Float16*)alloc((size_t)Lnum * 4 * 4096 * 2);
  _Float16* Wn1pk = (_Float16*)alloc((size_t)Lnum * 8 * 4096 * 2);
  _Float16* Wn2pk = (_Float16*)alloc((size_t)Lnum * 4 * 4096 * 2);
  _Float16* Wc1pk = (_Float16*)alloc((size_t)Lnum * 4 * 4096 * 2);
  (void)ws_size;

  auto cdiv = [](long a, long b) { return (int)((a + b - 1) / b); };

  // pack weights to WMMA-B fragment layout (f16, K-padded)
  egnn_pack_b<<<cdiv((long)Lnum*9*4096, 256), 256, 0, stream>>>(eW1, W1pk,  Lnum, 261, 9);
  egnn_pack_b<<<cdiv((long)Lnum*4*4096, 256), 256, 0, stream>>>(eW2, W2pk,  Lnum, 128, 4);
  egnn_pack_b<<<cdiv((long)Lnum*8*4096, 256), 256, 0, stream>>>(nW1, Wn1pk, Lnum, 256, 8);
  egnn_pack_b<<<cdiv((long)Lnum*4*4096, 256), 256, 0, stream>>>(nW2, Wn2pk, Lnum, 128, 4);
  egnn_pack_b<<<cdiv((long)Lnum*4*4096, 256), 256, 0, stream>>>(cW1, Wc1pk, Lnum, 128, 4);

  egnn_emb<<<cdiv((long)N * HID, 256), 256, 0, stream>>>(x_feat, emb_W, emb_b, h, N);
  hipMemcpyAsync(xb, pos, (size_t)N * 3 * 4, hipMemcpyDeviceToDevice, stream);
  egnn_zero<<<cdiv((long)G * HID, 256), 256, 0, stream>>>(sums, G * HID);
  egnn_zero<<<cdiv((long)G, 256), 256, 0, stream>>>(counts, G);

  const int nT32 = (E + 31) / 32;         // 32-edge tiles, 4 waves / 128-thread block
  const int nTn  = (N + 15) / 16;
  for (int l = 0; l < Lnum; ++l) {
    egnn_zero<<<cdiv((long)N * HID, 256), 256, 0, stream>>>(maggr, N * HID);
    egnn_zero<<<cdiv((long)N * 3, 256), 256, 0, stream>>>(xacc, N * 3);
    egnn_edge<<<cdiv(nT32, 4), 128, 0, stream>>>(
        h, xb, eattr, p_in, eidx, eidx + E, einv,
        (const v16h*)(W1pk + (size_t)l * 9 * 4096), eb1 + l * HID,
        (const v16h*)(W2pk + (size_t)l * 4 * 4096), eb2 + l * HID,
        gW + l * HID, gb + l,
        (const v16h*)(Wc1pk + (size_t)l * 4 * 4096), cb1 + l * HID,
        cW2 + l * HID, cb2 + l,
        maggr, xacc, nT32, E);
    egnn_axpy<<<cdiv((long)N * 3, 256), 256, 0, stream>>>(xb, xacc, N * 3);
    egnn_node<<<cdiv(nTn, 8), 256, 0, stream>>>(
        h, maggr,
        (const v16h*)(Wn1pk + (size_t)l * 8 * 4096), nb1 + l * HID,
        (const v16h*)(Wn2pk + (size_t)l * 4 * 4096), nb2 + l * HID,
        N, nTn);
  }
  egnn_pool<<<cdiv((long)N * HID, 256), 256, 0, stream>>>(h, batch, sums, counts, N);
  egnn_readout<<<cdiv((long)G, 256), 256, 0, stream>>>(sums, counts, rW, rb, (float*)d_out, G);
}
